// ModelArchitecture_70265664962947
// MI455X (gfx1250) — matrix operand order repaired and checked by reading, not compiled, not run on hardware
//
#include <hip/hip_runtime.h>
#include <hip/hip_bf16.h>
#include <cstdint>

// ---------------------------------------------------------------------------
// GATv2 x3 + BN + ELU + skip + dot-decode, MI455X (gfx1250, wave32, WMMA)
// Dense GEMMs: bf16 WMMA (v_wmma_f32_16x16x32_bf16), f32 accumulate.
// Weights stored transposed so both A and B WMMA fragments are K-contiguous
// in LDS (ds_load_b128 pairs, no strided u16 gathers).
// gfx1250 async global->LDS staging (ASYNCcnt) when the builtin is available.
// Edge softmax/aggregation: wave-per-edge, f32 atomics (ordered-uint max).
// ---------------------------------------------------------------------------

typedef __attribute__((ext_vector_type(16))) __bf16 v16bf;
typedef __attribute__((ext_vector_type(8)))  float  v8f;
typedef __attribute__((ext_vector_type(4)))  int    v4i;

#define GAT_N   50000
#define GAT_E   400000
#define GAT_EL  200000

// ------------------------- async global->LDS helpers -----------------------

#if defined(__HIP_DEVICE_COMPILE__) && \
    __has_builtin(__builtin_amdgcn_global_load_async_to_lds_b128)
#define ASYNC_LDS 1
__device__ __forceinline__ void cp16_async(void* l, const void* g) {
  __builtin_amdgcn_global_load_async_to_lds_b128(
      (__attribute__((address_space(1))) v4i*)(uintptr_t)g,
      (__attribute__((address_space(3))) v4i*)(uintptr_t)l, 0, 0);
}
__device__ __forceinline__ void async_wait0() {
#if __has_builtin(__builtin_amdgcn_s_wait_asynccnt)
  __builtin_amdgcn_s_wait_asynccnt(0);
#else
  asm volatile("s_wait_asynccnt 0x0" ::: "memory");
#endif
}
#endif

__device__ __forceinline__ void cp16(void* l, const void* g) {
  *(uint4*)l = *(const uint4*)g;
}

// ------------------------------ utility kernels ----------------------------

__global__ void k_f32_to_bf16(const float* __restrict__ in,
                              __bf16* __restrict__ out, int n) {
  int t = blockIdx.x * blockDim.x + threadIdx.x;
  if (t < n) out[t] = (__bf16)in[t];
}

// convert + transpose: in[K x Nc] f32 -> out[Nc x K] bf16
__global__ void k_f32_to_bf16_T(const float* __restrict__ in,
                                __bf16* __restrict__ out, int K, int Nc) {
  int t = blockIdx.x * blockDim.x + threadIdx.x;
  if (t >= K * Nc) return;
  int k = t / Nc, c = t - k * Nc;
  out[(size_t)c * K + k] = (__bf16)in[t];
}

__global__ void k_zero_u32(unsigned* __restrict__ p, int n) {
  int t = blockIdx.x * blockDim.x + threadIdx.x;
  if (t < n) p[t] = 0u;
}

// ------------------------------ WMMA GEMM ----------------------------------
// C[M x Nc] = A[M x K](bf16 rm) * Bt[Nc x K](bf16 rm, = B^T) + bias
// Block: 256 threads = 8 waves, tile BM=128 x BN=64, BK=32.
// Wave (wm 0..3, wn 0..1) owns a 32x32 sub-tile = 2x2 WMMA tiles.

#define BM 128
#define BN 64
#define BK 32
#define LDT_S (BK + 8)   // 40 bf16 = 80B rows (16B-aligned chunks)

__global__ __launch_bounds__(256) void k_gemm_bf16(
    const __bf16* __restrict__ A, const __bf16* __restrict__ Bt,
    const float* __restrict__ bias, float* __restrict__ C,
    int M, int K, int Nc) {
  __shared__ alignas(16) __bf16 As[BM][LDT_S];
  __shared__ alignas(16) __bf16 BsT[BN][LDT_S];

  const int tid  = threadIdx.x;
  const int lane = tid & 31;
  const int wid  = tid >> 5;
  const int wm   = wid & 3;   // 4 waves along M (32 rows each)
  const int wn   = wid >> 2;  // 2 waves along N (32 cols each)
  const int mblk = blockIdx.x * BM;
  const int nblk = blockIdx.y * BN;
  const bool interior = (mblk + BM) <= M;

  v8f acc[2][2] = {};

  for (int k0 = 0; k0 < K; k0 += BK) {
    // ---- stage A tile (BM x BK): 512 x 16B chunks, 2 per thread ----
    // ---- stage Bt tile (BN x BK): 256 x 16B chunks, 1 per thread ----
#ifdef ASYNC_LDS
    if (interior) {
#pragma unroll
      for (int it = 0; it < 2; ++it) {
        int idx = tid + it * 256;
        int r = idx >> 2, c8 = (idx & 3) * 8;
        cp16_async(&As[r][c8], A + (size_t)(mblk + r) * K + k0 + c8);
      }
      {
        int n = tid >> 2, c8 = (tid & 3) * 8;
        cp16_async(&BsT[n][c8], Bt + (size_t)(nblk + n) * K + k0 + c8);
      }
      async_wait0();
    } else
#endif
    {
#pragma unroll
      for (int it = 0; it < 2; ++it) {
        int idx = tid + it * 256;
        int r = idx >> 2, c8 = (idx & 3) * 8;
        int gr = mblk + r;
        if (gr < M) {
          cp16(&As[r][c8], A + (size_t)gr * K + k0 + c8);
        } else {
          uint4 z = {0, 0, 0, 0};
          *(uint4*)&As[r][c8] = z;
        }
      }
      {
        int n = tid >> 2, c8 = (tid & 3) * 8;
        cp16(&BsT[n][c8], Bt + (size_t)(nblk + n) * K + k0 + c8);
      }
    }
    __syncthreads();

    // ---- compute: 2x2 WMMA tiles per wave ----
#pragma unroll
    for (int i = 0; i < 2; ++i) {
      // A frag: lanes 0-15 -> M=lane, K{0..15}; lanes 16-31 K{16..31}
      v16bf a;
      const int m     = wm * 32 + i * 16 + (lane & 15);
      const int khalf = (lane >> 4) << 4;  // 0 or 16
#pragma unroll
      for (int e = 0; e < 16; ++e) {
        a[e]     = As[m][khalf + e];
      }
#pragma unroll
      for (int j = 0; j < 2; ++j) {
        // B frag: lane -> col n = lane&15; K = ((lane<16)?0:16)+e, contiguous
        v16bf b;
        const int n  = wn * 32 + j * 16 + (lane & 15);
        const int kb = (lane >> 4) << 4;   // 0 or 16
#pragma unroll
        for (int e = 0; e < 16; ++e) b[e] = BsT[n][kb + e];

        acc[i][j] = __builtin_amdgcn_wmma_f32_16x16x32_bf16(
            false, a, false, b, (short)0, acc[i][j], false, false);
      }
    }
    __syncthreads();
  }

  // ---- store C (+bias): lane -> n = lane&15; vgpr r -> m = r + 8*(lane>=16)
#pragma unroll
  for (int i = 0; i < 2; ++i) {
#pragma unroll
    for (int j = 0; j < 2; ++j) {
      const int n  = nblk + wn * 32 + j * 16 + (lane & 15);
      const float bv = bias ? bias[n] : 0.0f;
#pragma unroll
      for (int r = 0; r < 8; ++r) {
        const int m = mblk + wm * 32 + i * 16 + r + ((lane >> 4) << 3);
        if (m < M) C[(size_t)m * Nc + n] = acc[i][j][r] + bv;
      }
    }
  }
}

// ------------------------- edge-phase kernels ------------------------------

__device__ __forceinline__ unsigned f2ord(float f) {
  unsigned u = __float_as_uint(f);
  return (u & 0x80000000u) ? ~u : (u | 0x80000000u);
}
__device__ __forceinline__ float ord2f(unsigned u) {
  return (u & 0x80000000u) ? __uint_as_float(u ^ 0x80000000u)
                           : __uint_as_float(~u);
}

// Pass 1 (wave per edge*head): logit = att . leaky_relu(xl[src]+xr[dst]);
// segment max into mbuf.
__global__ void k_edge_logit(const float* __restrict__ xl,
                             const float* __restrict__ xr,
                             const float* __restrict__ att,
                             const int* __restrict__ src,
                             const int* __restrict__ dst,
                             float* __restrict__ logits,
                             unsigned* __restrict__ mbuf,
                             int E, int H, int Cc) {
  int w    = (blockIdx.x * blockDim.x + threadIdx.x) >> 5;
  int lane = threadIdx.x & 31;
  if (w >= E * H) return;
  int e = w / H, h = w - e * H;
  int s = src[e], d = dst[e];
  const float* pl = xl + (size_t)s * H * Cc + h * Cc;
  const float* pr = xr + (size_t)d * H * Cc + h * Cc;
  const float* pa = att + h * Cc;
  float acc = 0.0f;
  for (int c = lane; c < Cc; c += 32) {
    float v = pl[c] + pr[c];
    v = (v > 0.0f) ? v : 0.2f * v;
    acc += v * pa[c];
  }
#pragma unroll
  for (int off = 16; off > 0; off >>= 1) acc += __shfl_xor(acc, off, 32);
  if (lane == 0) {
    logits[w] = acc;
    atomicMax(&mbuf[(size_t)d * H + h], f2ord(acc));
  }
}

// Pass 2 (thread per edge*head): ex = exp(logit - max[dst]); sum into denom
__global__ void k_edge_exp(float* __restrict__ logits,
                           const unsigned* __restrict__ mbuf,
                           float* __restrict__ denom,
                           const int* __restrict__ dst, int E, int H) {
  int t = blockIdx.x * blockDim.x + threadIdx.x;
  if (t >= E * H) return;
  int e = t / H, h = t - e * H;
  int d = dst[e];
  float m  = ord2f(mbuf[(size_t)d * H + h]);
  float ex = __expf(logits[t] - m);
  logits[t] = ex;
  atomicAdd(&denom[(size_t)d * H + h], ex);
}

// Pass 3 (wave per edge*head): agg[dst] += alpha * xl[src]
__global__ void k_edge_agg(const float* __restrict__ ex,
                           const float* __restrict__ denom,
                           const float* __restrict__ xl,
                           const int* __restrict__ src,
                           const int* __restrict__ dst,
                           float* __restrict__ agg, int E, int H, int Cc) {
  int w    = (blockIdx.x * blockDim.x + threadIdx.x) >> 5;
  int lane = threadIdx.x & 31;
  if (w >= E * H) return;
  int e = w / H, h = w - e * H;
  int s = src[e], d = dst[e];
  float alpha = ex[w] / (denom[(size_t)d * H + h] + 1e-16f);
  const float* pl = xl  + (size_t)s * H * Cc + h * Cc;
  float*       pg = agg + (size_t)d * H * Cc + h * Cc;
  for (int c = lane; c < Cc; c += 32) atomicAdd(&pg[c], alpha * pl[c]);
}

// ------------------------------ BatchNorm ----------------------------------
// (per-channel GAT bias cancels under BN, so it is skipped entirely)

__global__ void k_bn_stats(const float* __restrict__ X,
                           float* __restrict__ sum, float* __restrict__ sumsq,
                           int nrows, int C) {
  int c  = threadIdx.x;  // blockDim.x == C
  int r0 = blockIdx.x * 256;
  int r1 = r0 + 256; if (r1 > nrows) r1 = nrows;
  float s = 0.0f, s2 = 0.0f;
  for (int r = r0; r < r1; ++r) {
    float v = X[(size_t)r * C + c];
    s += v; s2 += v * v;
  }
  atomicAdd(&sum[c], s);
  atomicAdd(&sumsq[c], s2);
}

__global__ void k_bn_elu_bf16(const float* __restrict__ X,
                              const float* __restrict__ sum,
                              const float* __restrict__ sumsq,
                              const float* __restrict__ g,
                              const float* __restrict__ b,
                              __bf16* __restrict__ out, int nrows, int C) {
  int t = blockIdx.x * blockDim.x + threadIdx.x;
  if (t >= nrows * C) return;
  int c = t % C;
  float inv = 1.0f / (float)nrows;
  float mu  = sum[c] * inv;
  float var = sumsq[c] * inv - mu * mu;
  float y = g[c] * (X[t] - mu) * rsqrtf(var + 1e-5f) + b[c];
  y = (y > 0.0f) ? y : (__expf(y) - 1.0f);   // ELU
  out[t] = (__bf16)y;
}

// --------------------------- final combine + decode ------------------------

__global__ void k_add3(const float* __restrict__ agg,
                       const float* __restrict__ bias,
                       const float* __restrict__ ident,
                       float* __restrict__ z, int nrows, int C) {
  int t = blockIdx.x * blockDim.x + threadIdx.x;
  if (t >= nrows * C) return;
  z[t] = agg[t] + bias[t % C] + ident[t];
}

// one wave (32 lanes) per label edge: dot(z[a], z[b]) over 128 channels
__global__ void k_decode(const float* __restrict__ z,
                         const int* __restrict__ el0,
                         const int* __restrict__ el1,
                         float* __restrict__ out, int EL) {
  int w    = (blockIdx.x * blockDim.x + threadIdx.x) >> 5;
  int lane = threadIdx.x & 31;
  if (w >= EL) return;
  int a = el0[w], b = el1[w];
  const float* za = z + (size_t)a * 128;
  const float* zb = z + (size_t)b * 128;
  float s = 0.0f;
  for (int c = lane; c < 128; c += 32) s += za[c] * zb[c];
#pragma unroll
  for (int off = 16; off > 0; off >>= 1) s += __shfl_xor(s, off, 32);
  if (lane == 0) out[w] = s;
}

// ------------------------------ host driver --------------------------------

static inline int cdiv(int a, int b) { return (a + b - 1) / b; }

extern "C" void kernel_launch(void* const* d_in, const int* in_sizes, int n_in,
                              void* d_out, int out_size, void* d_ws, size_t ws_size,
                              hipStream_t stream) {
  const float* x   = (const float*)d_in[0];
  const int*   ei  = (const int*)d_in[1];   // [2, E]
  const int*   el  = (const int*)d_in[2];   // [2, EL]
  const float* Wl1 = (const float*)d_in[3];
  const float* bl1 = (const float*)d_in[4];
  const float* Wr1 = (const float*)d_in[5];
  const float* br1 = (const float*)d_in[6];
  const float* at1 = (const float*)d_in[7];
  const float* Wl2 = (const float*)d_in[9];
  const float* bl2 = (const float*)d_in[10];
  const float* Wr2 = (const float*)d_in[11];
  const float* br2 = (const float*)d_in[12];
  const float* at2 = (const float*)d_in[13];
  const float* Wl3 = (const float*)d_in[15];
  const float* bl3 = (const float*)d_in[16];
  const float* Wr3 = (const float*)d_in[17];
  const float* br3 = (const float*)d_in[18];
  const float* at3 = (const float*)d_in[19];
  const float* bi3 = (const float*)d_in[20];
  const float* g1  = (const float*)d_in[21];
  const float* be1 = (const float*)d_in[22];
  const float* g2  = (const float*)d_in[23];
  const float* be2 = (const float*)d_in[24];
  const float* Ws  = (const float*)d_in[25];
  const float* bs  = (const float*)d_in[26];

  const int N = GAT_N, E = GAT_E, EL = GAT_EL;
  const int* src = ei;
  const int* dst = ei + E;
  const int* el0 = el;
  const int* el1 = el + EL;

  // ---------------- workspace carve (256B aligned) ----------------
  char* base = (char*)d_ws;
  size_t off = 0;
  auto carve = [&](size_t bytes) -> char* {
    char* p = base + off;
    off += (bytes + 255) & ~(size_t)255;
    return p;
  };
  __bf16* xb    = (__bf16*)carve((size_t)N * 128 * 2);
  __bf16* wb_l1 = (__bf16*)carve(128 * 256 * 2);   // transposed: [256 x 128]
  __bf16* wb_r1 = (__bf16*)carve(128 * 256 * 2);
  __bf16* wb_l2 = (__bf16*)carve(256 * 256 * 2);
  __bf16* wb_r2 = (__bf16*)carve(256 * 256 * 2);
  __bf16* wb_l3 = (__bf16*)carve(256 * 128 * 2);
  __bf16* wb_r3 = (__bf16*)carve(256 * 128 * 2);
  __bf16* wb_s  = (__bf16*)carve(128 * 128 * 2);
  float*  xlb  = (float*)carve((size_t)N * 256 * 4);
  float*  xrb  = (float*)carve((size_t)N * 256 * 4);
  float*  agg  = (float*)carve((size_t)N * 256 * 4);
  __bf16* hb1  = (__bf16*)carve((size_t)N * 256 * 2);
  __bf16* hb2  = (__bf16*)carve((size_t)N * 256 * 2);
  float*  iden = (float*)carve((size_t)N * 128 * 4);
  float*  zbuf = (float*)carve((size_t)N * 128 * 4);
  float*  lgts = (float*)carve((size_t)E * 4 * 4);
  unsigned* mbuf = (unsigned*)carve((size_t)N * 4 * 4);
  float*  dnm  = (float*)carve((size_t)N * 4 * 4);
  float*  bsum = (float*)carve(256 * 4);
  float*  bsq  = (float*)carve(256 * 4);
  (void)ws_size; (void)n_in; (void)in_sizes; (void)out_size;

  const int TB = 256;

  // ---------------- convert input + transposed bf16 weights ----------------
  k_f32_to_bf16<<<cdiv(N * 128, TB), TB, 0, stream>>>(x, xb, N * 128);
  k_f32_to_bf16_T<<<cdiv(128 * 256, TB), TB, 0, stream>>>(Wl1, wb_l1, 128, 256);
  k_f32_to_bf16_T<<<cdiv(128 * 256, TB), TB, 0, stream>>>(Wr1, wb_r1, 128, 256);
  k_f32_to_bf16_T<<<cdiv(256 * 256, TB), TB, 0, stream>>>(Wl2, wb_l2, 256, 256);
  k_f32_to_bf16_T<<<cdiv(256 * 256, TB), TB, 0, stream>>>(Wr2, wb_r2, 256, 256);
  k_f32_to_bf16_T<<<cdiv(256 * 128, TB), TB, 0, stream>>>(Wl3, wb_l3, 256, 128);
  k_f32_to_bf16_T<<<cdiv(256 * 128, TB), TB, 0, stream>>>(Wr3, wb_r3, 256, 128);
  k_f32_to_bf16_T<<<cdiv(128 * 128, TB), TB, 0, stream>>>(Ws,  wb_s,  128, 128);

  // ---------------- identity = x @ Ws + bs ----------------
  {
    dim3 g(cdiv(N, BM), 128 / BN);
    k_gemm_bf16<<<g, TB, 0, stream>>>(xb, wb_s, bs, iden, N, 128, 128);
  }

  // ============ generic GAT layer runner (host-side lambda) ============
  auto run_gat = [&](const __bf16* inp, int K, const __bf16* wl, const float* bl,
                     const __bf16* wr, const float* br, const float* att,
                     int H, int Cc) {
    const int HC = H * Cc;
    dim3 g(cdiv(N, BM), HC / BN);
    k_gemm_bf16<<<g, TB, 0, stream>>>(inp, wl, bl, xlb, N, K, HC);
    k_gemm_bf16<<<g, TB, 0, stream>>>(inp, wr, br, xrb, N, K, HC);
    k_zero_u32<<<cdiv(N * H, TB), TB, 0, stream>>>(mbuf, N * H);
    k_zero_u32<<<cdiv(N * H, TB), TB, 0, stream>>>((unsigned*)dnm, N * H);
    k_zero_u32<<<cdiv(N * HC, TB), TB, 0, stream>>>((unsigned*)agg, N * HC);
    const int EH = E * H;
    k_edge_logit<<<cdiv(EH * 32, TB), TB, 0, stream>>>(xlb, xrb, att, src, dst,
                                                       lgts, mbuf, E, H, Cc);
    k_edge_exp<<<cdiv(EH, TB), TB, 0, stream>>>(lgts, mbuf, dnm, dst, E, H);
    k_edge_agg<<<cdiv(EH * 32, TB), TB, 0, stream>>>(lgts, dnm, xlb, src, dst,
                                                     agg, E, H, Cc);
  };

  // ---------------- layer 1: 128 -> 4x64, BN + ELU ----------------
  run_gat(xb, 128, wb_l1, bl1, wb_r1, br1, at1, 4, 64);
  k_zero_u32<<<2, TB, 0, stream>>>((unsigned*)bsum, 256);
  k_zero_u32<<<2, TB, 0, stream>>>((unsigned*)bsq, 256);
  k_bn_stats<<<cdiv(N, 256), 256, 0, stream>>>(agg, bsum, bsq, N, 256);
  k_bn_elu_bf16<<<cdiv(N * 256, TB), TB, 0, stream>>>(agg, bsum, bsq, g1, be1,
                                                      hb1, N, 256);

  // ---------------- layer 2: 256 -> 4x64, BN + ELU ----------------
  run_gat(hb1, 256, wb_l2, bl2, wb_r2, br2, at2, 4, 64);
  k_zero_u32<<<2, TB, 0, stream>>>((unsigned*)bsum, 256);
  k_zero_u32<<<2, TB, 0, stream>>>((unsigned*)bsq, 256);
  k_bn_stats<<<cdiv(N, 256), 256, 0, stream>>>(agg, bsum, bsq, N, 256);
  k_bn_elu_bf16<<<cdiv(N * 256, TB), TB, 0, stream>>>(agg, bsum, bsq, g2, be2,
                                                      hb2, N, 256);

  // ---------------- layer 3: 256 -> 1x128 (mean over 1 head == identity) --
  run_gat(hb2, 256, wb_l3, bl3, wb_r3, br3, at3, 1, 128);

  // z = agg3 + bias3 + identity
  k_add3<<<cdiv(N * 128, TB), TB, 0, stream>>>(agg, bi3, iden, zbuf, N, 128);

  // decode: out[e] = dot(z[el0[e]], z[el1[e]])
  k_decode<<<cdiv(EL * 32, TB), TB, 0, stream>>>(zbuf, el0, el1,
                                                 (float*)d_out, EL);
}
